// MyLSTM_11244224381643
// MI455X (gfx1250) — matrix operand order repaired and checked by reading, not compiled, not run on hardware
//
#include <hip/hip_runtime.h>

#define BATCH 64
#define SEQ   2048
#define ISZ   256
#define HSZ   256

typedef __attribute__((ext_vector_type(8)))  __bf16 v8bf;
typedef __attribute__((ext_vector_type(16))) __bf16 v16bf;
typedef __attribute__((ext_vector_type(8)))  float  v8f;

static __device__ __forceinline__ unsigned short f2bf(float f) {
    unsigned int u = __float_as_uint(f);
    unsigned int r = (u + 0x7FFFu + ((u >> 16) & 1u)) >> 16;
    return (unsigned short)r;
}

static __device__ __forceinline__ v16bf cat16(v8bf lo, v8bf hi) {
    return __builtin_shufflevector(lo, hi, 0,1,2,3,4,5,6,7,8,9,10,11,12,13,14,15);
}

// ---------------------------------------------------------------------------
// Prepack: W = [U;V] (512 x 1024 fp32) -> bf16, grouped per hidden-group WG.
// Wpack layout: [hg=8][r=128][k=512] bf16, r = gate*32 + local hidden unit.
// ---------------------------------------------------------------------------
__global__ void lstm_prepack(const float* __restrict__ Ui, const float* __restrict__ Vi,
                             const float* __restrict__ Uf, const float* __restrict__ Vf,
                             const float* __restrict__ Uo, const float* __restrict__ Vo,
                             const float* __restrict__ Uc, const float* __restrict__ Vc,
                             unsigned short* __restrict__ Wpack) {
    int e = blockIdx.x * blockDim.x + threadIdx.x;    // 8*128*512 = 524288
    if (e >= 8 * 128 * 512) return;
    int hg  = e >> 16;
    int rem = e & 65535;
    int r   = rem >> 9;
    int k   = rem & 511;
    int g   = r >> 5;
    int loc = r & 31;
    int col = hg * 32 + loc;
    const float* U = (g == 0) ? Ui : (g == 1) ? Uf : (g == 2) ? Uo : Uc;
    const float* V = (g == 0) ? Vi : (g == 1) ? Vf : (g == 2) ? Vo : Vc;
    float val = (k < 256) ? U[k * HSZ + col] : V[(k - 256) * HSZ + col];
    Wpack[e] = f2bf(val);
}

// Zero h double-buffer (bf16: bit-zero == 0.0) and barrier counters.
__global__ void lstm_init(unsigned short* __restrict__ hbuf, unsigned int* __restrict__ counters) {
    int i = blockIdx.x * blockDim.x + threadIdx.x;
    if (i < 2 * BATCH * HSZ) hbuf[i] = 0;
    if (i < 4) counters[i] = 0u;
}

// ---------------------------------------------------------------------------
// Persistent recurrent kernel. 32 WGs = 4 batch groups (16 rows) x 8 hidden
// groups (128 gate-columns). Weights resident in LDS (bf16, 133 KB). Per step:
//   z = [x_t | h] @ Wslice via v_wmma_f32_16x16x32_bf16 (16 K-tiles, 2 acc
//   chains), gates -> c,h update, h broadcast in bf16 via L2 + atomic barrier.
// x-half of step t+1 is staged BEFORE the inter-WG barrier (off critical path);
// only the 8KB h-half is staged after it.
// ---------------------------------------------------------------------------
__global__ void __launch_bounds__(256, 1)
lstm_recurrent(const float* __restrict__ x,
               const float* __restrict__ bi, const float* __restrict__ bfp,
               const float* __restrict__ bo, const float* __restrict__ bc,
               const unsigned short* __restrict__ Wpack,
               unsigned short* __restrict__ hbuf,   // [2][64][256] bf16
               unsigned int* __restrict__ counters, // [4]
               float* __restrict__ out) {
    __shared__ unsigned short Ws[128 * 520];   // B slice, pitch-padded (133 KB)
    __shared__ unsigned short As[16 * 520];    // [x_t | h] bf16 staging
    __shared__ float zs[16 * 128];             // z tile for cross-wave gate mix
    __shared__ float cs[512];                  // cell state [16 rows][32 units]
    __shared__ float bias_s[128];

    const int tid  = threadIdx.x;
    const int bg   = blockIdx.x >> 3;          // batch group 0..3
    const int hg   = blockIdx.x & 7;           // hidden group 0..7
    const int wave = tid >> 5;                 // N-tile 0..7
    const int lane = tid & 31;
    const int lrow = lane & 15;
    const int hi   = lane >> 4;

    // Load resident weight slice into LDS.
    const unsigned short* wp = Wpack + (size_t)hg * 65536;
    for (int e = tid; e < 128 * 512; e += 256) {
        int r = e >> 9, k = e & 511;
        Ws[r * 520 + k] = wp[e];
    }
    if (tid < 128) {
        int g = tid >> 5, n = tid & 31;
        const float* bp = (g == 0) ? bi : (g == 1) ? bfp : (g == 2) ? bo : bc;
        bias_s[tid] = bp[hg * 32 + n];
    }
    for (int e = tid; e < 512; e += 256) cs[e] = 0.0f;

    // Prologue: stage x-half for t=0.
    for (int e = tid; e < 16 * 256; e += 256) {
        int m = e >> 8, k = e & 255;
        int b = bg * 16 + m;
        const float* px = x + ((size_t)b * SEQ) * ISZ + k;
        As[m * 520 + k] = f2bf(*px);
        if ((k & 31) == 0) __builtin_prefetch(px + ISZ, 0, 3);
    }
    __syncthreads();

    volatile unsigned int* cnt = counters + bg;
    const size_t HS_TOTAL = (size_t)BATCH * SEQ * HSZ;

    for (int t = 0; t < SEQ; ++t) {
        // ---- Stage h-half (bf16 copy, no convert) from broadcast buffer ----
        const unsigned short* hread = hbuf + ((t + 1) & 1) * (BATCH * HSZ);
        for (int e = tid; e < 16 * 256; e += 256) {
            int m = e >> 8, k = e & 255;
            As[m * 520 + 256 + k] = hread[(bg * 16 + m) * HSZ + k];
        }
        __syncthreads();

        // ---- GEMM: each wave owns one 16-col N-tile; 2 accumulator chains ----
        v8f acc0 = {}, acc1 = {};
        #pragma unroll 2
        for (int kt = 0; kt < 16; kt += 2) {
            {
                const int ak = kt * 32 + (hi ? 8 : 0);
                v8bf a_lo = *(const v8bf*)&As[lrow * 520 + ak];
                v8bf a_hi = *(const v8bf*)&As[lrow * 520 + ak + 16];
                const int bk = kt * 32 + hi * 8;
                v8bf b_lo = *(const v8bf*)&Ws[(wave * 16 + lrow) * 520 + bk];
                v8bf b_hi = *(const v8bf*)&Ws[(wave * 16 + lrow) * 520 + bk + 16];
                acc0 = __builtin_amdgcn_wmma_f32_16x16x32_bf16(
                    false, cat16(a_lo, a_hi), false, cat16(b_lo, b_hi),
                    (short)0, acc0, false, false);
            }
            {
                const int ak = (kt + 1) * 32 + (hi ? 8 : 0);
                v8bf a_lo = *(const v8bf*)&As[lrow * 520 + ak];
                v8bf a_hi = *(const v8bf*)&As[lrow * 520 + ak + 16];
                const int bk = (kt + 1) * 32 + hi * 8;
                v8bf b_lo = *(const v8bf*)&Ws[(wave * 16 + lrow) * 520 + bk];
                v8bf b_hi = *(const v8bf*)&Ws[(wave * 16 + lrow) * 520 + bk + 16];
                acc1 = __builtin_amdgcn_wmma_f32_16x16x32_bf16(
                    false, cat16(a_lo, a_hi), false, cat16(b_lo, b_hi),
                    (short)0, acc1, false, false);
            }
        }
        #pragma unroll
        for (int r = 0; r < 8; ++r)
            zs[(r + hi * 8) * 128 + wave * 16 + lrow] = acc0[r] + acc1[r];
        __syncthreads();   // also guarantees all waves are done reading As

        // ---- Gates + state update (512 values, 2 per thread) ----
        unsigned short* hwrite = hbuf + (t & 1) * (BATCH * HSZ);
        for (int e = tid; e < 512; e += 256) {
            int m = e >> 5, n = e & 31;
            float zi = zs[m * 128 + n]      + bias_s[n];
            float zf = zs[m * 128 + 32 + n] + bias_s[32 + n];
            float zo = zs[m * 128 + 64 + n] + bias_s[64 + n];
            float zc = zs[m * 128 + 96 + n] + bias_s[96 + n];
            float ig = 1.0f / (1.0f + __expf(-zi));
            float fg = 1.0f / (1.0f + __expf(-zf));
            float og = 1.0f / (1.0f + __expf(-zo));
            float gg = 2.0f / (1.0f + __expf(-2.0f * zc)) - 1.0f;
            float cn = fg * cs[e] + ig * gg;
            cs[e] = cn;
            float hn = og * (2.0f / (1.0f + __expf(-2.0f * cn)) - 1.0f);
            int b = bg * 16 + m;
            int col = hg * 32 + n;
            hwrite[b * HSZ + col] = f2bf(hn);          // bf16 broadcast copy
            out[((size_t)b * SEQ + t) * HSZ + col] = hn;
            if (t == SEQ - 1) {
                out[HS_TOTAL + (size_t)b * HSZ + col] = hn;                       // h_T
                out[HS_TOTAL + (size_t)BATCH * HSZ + (size_t)b * HSZ + col] = cn; // c_T
            }
        }

        // ---- Stage x-half for t+1 (overlaps barrier wait; As free post-sync) ----
        if (t + 1 < SEQ) {
            for (int e = tid; e < 16 * 256; e += 256) {
                int m = e >> 8, k = e & 255;
                int b = bg * 16 + m;
                const float* px = x + ((size_t)b * SEQ + (t + 1)) * ISZ + k;
                As[m * 520 + k] = f2bf(*px);
                if (((k & 31) == 0) && (t + 2 < SEQ))
                    __builtin_prefetch(px + ISZ, 0, 3);   // global_prefetch_b8
            }
        }

        // ---- Release/acquire barrier across the 8 WGs of this batch group ----
        __threadfence();
        __syncthreads();
        if (tid == 0) {
            atomicAdd((unsigned int*)cnt, 1u);
            unsigned int target = (unsigned int)(8 * (t + 1));
            while (*cnt < target) __builtin_amdgcn_s_sleep(1);
            __threadfence();
        }
        __syncthreads();
    }
}

extern "C" void kernel_launch(void* const* d_in, const int* in_sizes, int n_in,
                              void* d_out, int out_size, void* d_ws, size_t ws_size,
                              hipStream_t stream) {
    const float* x  = (const float*)d_in[0];
    const float* Ui = (const float*)d_in[1];
    const float* Vi = (const float*)d_in[2];
    const float* bi = (const float*)d_in[3];
    const float* Uf = (const float*)d_in[4];
    const float* Vf = (const float*)d_in[5];
    const float* bf = (const float*)d_in[6];
    const float* Uo = (const float*)d_in[7];
    const float* Vo = (const float*)d_in[8];
    const float* bo = (const float*)d_in[9];
    const float* Uc = (const float*)d_in[10];
    const float* Vc = (const float*)d_in[11];
    const float* bc = (const float*)d_in[12];

    // Workspace: [0, 1MB) bf16 packed weights; then bf16 h double buffer (64KB);
    // then 4 barrier counters.
    unsigned short* Wpack    = (unsigned short*)d_ws;
    unsigned short* hbuf     = (unsigned short*)((char*)d_ws + (1u << 20));
    unsigned int*   counters = (unsigned int*)((char*)d_ws + (1u << 20) +
                                               2 * BATCH * HSZ * sizeof(unsigned short));
    float*          out      = (float*)d_out;

    lstm_prepack<<<2048, 256, 0, stream>>>(Ui, Vi, Uf, Vf, Uo, Vo, Uc, Vc, Wpack);
    lstm_init<<<128, 256, 0, stream>>>(hbuf, counters);
    lstm_recurrent<<<32, 256, 0, stream>>>(x, bi, bf, bo, bc, Wpack, hbuf, counters, out);
}
